// MessagePassing_80719615361613
// MI455X (gfx1250) — matrix operand order repaired and checked by reading, not compiled, not run on hardware
//
#include <hip/hip_runtime.h>

// Problem constants (match reference)
#define Cc 128
#define Kk 9
#define KDIM (Cc * Kk)          // 1152 = im2col K dimension
#define Bb 8
#define Hh 128
#define Ww 256
#define STRIDE_B (Cc * Hh * Ww) // 4194304 elements per batch
#define WELEMS (Cc * KDIM)      // 147456 elements per weight tensor
#define BPITCH 1160             // padded LDS pitch (elems): 2320B, 16B-aligned rows, no bank clash

typedef __attribute__((ext_vector_type(16))) __bf16 v16bf;
typedef __attribute__((ext_vector_type(8)))  __bf16 v8bf;
typedef __attribute__((ext_vector_type(8)))  float  v8f;

// Convert the four fp32 weight tensors (C,C,K) to bf16 im2col A-matrices [co][ci*9+k]
__global__ __launch_bounds__(256)
void mp_wconv_kernel(const float* __restrict__ w0, const float* __restrict__ w1,
                     const float* __restrict__ w2, const float* __restrict__ w3,
                     __bf16* __restrict__ o) {
    int i = blockIdx.x * 256 + threadIdx.x;
    if (i < WELEMS) {
        o[0 * WELEMS + i] = (__bf16)w0[i];
        o[1 * WELEMS + i] = (__bf16)w1[i];
        o[2 * WELEMS + i] = (__bf16)w2[i];
        o[3 * WELEMS + i] = (__bf16)w3[i];
    }
}

// One recurrence step: cur_line += relu( W (*) prev_line ), conv along axis of length L.
// Element (b, c, l) of a line lives at state + off + b*STRIDE_B + c*strideC + l*strideL.
// Grid: (L/16, B); block: 256 threads = 8 waves; wave w owns output rows [16w, 16w+16).
__global__ __launch_bounds__(256)
void mp_step_kernel(float* __restrict__ state, const __bf16* __restrict__ Wm,
                    long long offCur, long long offPrev,
                    int L, int strideC, int strideL) {
    __shared__ __bf16 Pch[Cc * 24];                       // halo chunk of prev line (6 KB)
    __shared__ __align__(16) __bf16 Bm[16 * BPITCH];      // im2col B matrix (36.3 KB)

    const int tid = threadIdx.x;
    const int b   = blockIdx.y;
    const int n0  = blockIdx.x * 16;

    const float* prev = state + offPrev + (long long)b * STRIDE_B;
    float*       cur  = state + offCur  + (long long)b * STRIDE_B;

    // Stage 1: gather prev-line halo [128 ci][24 cols] -> LDS bf16 (zero-padded ends)
    for (int idx = tid; idx < Cc * 24; idx += 256) {
        int ci = idx / 24, c = idx % 24;
        int l  = n0 - 4 + c;
        float v = 0.f;
        if (l >= 0 && l < L) v = prev[(long long)ci * strideC + (long long)l * strideL];
        Pch[ci * 24 + c] = (__bf16)v;
    }
    __syncthreads();

    // Stage 2: im2col expand: Bm[n][kk] = Pch[ci][n + k], kk = ci*9 + k
    for (int idx = tid; idx < 16 * KDIM; idx += 256) {
        int n  = idx / KDIM;
        int kk = idx - n * KDIM;
        int ci = kk / Kk;
        int k  = kk - ci * Kk;
        Bm[n * BPITCH + kk] = Pch[ci * 24 + n + k];
    }
    __syncthreads();

    // Stage 3: WMMA. Wave -> 16x16 output tile (M = wave*16.., N = n0..n0+15), K-loop 1152.
    const int  lane  = tid & 31;
    const int  wv    = tid >> 5;
    const int  m0    = wv * 16;
    const int  nl    = lane & 15;
    const bool hiLn  = lane >= 16;
    const int  koffB = hiLn ? 8 : 0;   // B fragment: upper lanes hold K 8..15 / 24..31
    const int  khA   = hiLn ? 8 : 0;   // A fragment: upper lanes hold K 8..15 / 24..31

    const __bf16* Arow = Wm + (long long)(m0 + nl) * KDIM; // A row M = m0 + (lane&15)
    const __bf16* Brow = Bm + nl * BPITCH;                 // B column N = lane&15

    v8f acc = {};
    #pragma unroll 4
    for (int kb = 0; kb < KDIM; kb += 32) {
        v8bf alo = *(const v8bf*)(Arow + kb + khA);        // K = kb+khA .. +7
        v8bf ahi = *(const v8bf*)(Arow + kb + 16 + khA);   // K = kb+16+khA .. +7
        v8bf blo = *(const v8bf*)(Brow + kb + koffB);      // K = kb+koffB .. +7
        v8bf bhi = *(const v8bf*)(Brow + kb + koffB + 16); // K = kb+16+koffB .. +7
        v16bf a  = __builtin_shufflevector(alo, ahi, 0,1,2,3,4,5,6,7,8,9,10,11,12,13,14,15);
        v16bf bt = __builtin_shufflevector(blo, bhi, 0,1,2,3,4,5,6,7,8,9,10,11,12,13,14,15);
        acc = __builtin_amdgcn_wmma_f32_16x16x32_bf16(
                  /*neg_a=*/false, a, /*neg_b=*/false, bt,
                  /*c_mod=*/(short)0, acc, /*reuse_a=*/false, /*reuse_b=*/false);
    }

    // D layout: N = lane&15; VGPR r -> M = r (lanes 0-15) or 8+r (lanes 16-31)
    const int mbase = m0 + (hiLn ? 8 : 0);
    const int l     = n0 + nl;
    #pragma unroll
    for (int r = 0; r < 8; r++) {
        float v = acc[r];
        v = v > 0.f ? v : 0.f;
        float* p = cur + (long long)(mbase + r) * strideC + (long long)l * strideL;
        *p += v;
    }
}

extern "C" void kernel_launch(void* const* d_in, const int* in_sizes, int n_in,
                              void* d_out, int out_size, void* d_ws, size_t ws_size,
                              hipStream_t stream) {
    const float* x  = (const float*)d_in[0];
    const float* wd = (const float*)d_in[1];
    const float* wu = (const float*)d_in[2];
    const float* wr = (const float*)d_in[3];
    const float* wl = (const float*)d_in[4];
    float* out = (float*)d_out;
    __bf16* Wmat = (__bf16*)d_ws;   // 4 * 147456 bf16 = 1.18 MB

    // Weights -> bf16 im2col matrices in workspace
    mp_wconv_kernel<<<(WELEMS + 255) / 256, 256, 0, stream>>>(wd, wu, wr, wl, Wmat);

    // State starts as x; all four passes update it in place
    hipMemcpyAsync(out, x, (size_t)Bb * STRIDE_B * sizeof(float),
                   hipMemcpyDeviceToDevice, stream);

    const dim3 blk(256);
    const dim3 gridH(Ww / 16, Bb);  // conv along W (down/up): 16 tiles x 8 batches
    const dim3 gridW(Hh / 16, Bb);  // conv along H (right/left): 8 tiles x 8 batches
    const int sC = Hh * Ww;         // channel stride

    // Pass 1: top -> bottom (prop along H, conv along W)
    for (int i = 1; i < Hh; i++)
        mp_step_kernel<<<gridH, blk, 0, stream>>>(out, Wmat + 0 * WELEMS,
            (long long)i * Ww, (long long)(i - 1) * Ww, Ww, sC, 1);
    // Pass 2: bottom -> top
    for (int i = Hh - 2; i >= 0; i--)
        mp_step_kernel<<<gridH, blk, 0, stream>>>(out, Wmat + 1 * WELEMS,
            (long long)i * Ww, (long long)(i + 1) * Ww, Ww, sC, 1);
    // Pass 3: left -> right (prop along W, conv along H)
    for (int i = 1; i < Ww; i++)
        mp_step_kernel<<<gridW, blk, 0, stream>>>(out, Wmat + 2 * WELEMS,
            (long long)i, (long long)(i - 1), Hh, sC, Ww);
    // Pass 4: right -> left
    for (int i = Ww - 2; i >= 0; i--)
        mp_step_kernel<<<gridW, blk, 0, stream>>>(out, Wmat + 3 * WELEMS,
            (long long)i, (long long)(i + 1), Hh, sC, Ww);
}
